// LatentMutator_19146964206406
// MI455X (gfx1250) — compile-verified, matrix operand order checked by reading, not run on hardware
//
#include <hip/hip_runtime.h>
#include <math.h>

#define N_ATTR 12
#define N_CH   64
#define N_LEV  3
#define N_B    16
#define IN_DIM 219
#define IN_PAD 224
#define H1     256
#define H2     128
#define NHEAD  36   /* A*L */

typedef __attribute__((ext_vector_type(16))) _Float16 v16h;
typedef __attribute__((ext_vector_type(8)))  float    v8f;

// CDNA5 16-bit A/B fragment K index for 16x16x32: lane holds row/col (lane&15),
// element e -> K = (lane>=16?8:0) + (e&7) + (e>=8?16:0)
__device__ __forceinline__ int wmma_k_idx(int lane, int e) {
    return ((lane >> 4) << 3) + (e & 7) + ((e & 8) << 1);
}

// ---------------- Kernel 1: spatial mean pooling of z and ig ----------------
__global__ void pool_kernel(const float* __restrict__ z0, const float* __restrict__ z1,
                            const float* __restrict__ z2, const float* __restrict__ ig0,
                            const float* __restrict__ ig1, const float* __restrict__ ig2,
                            float* __restrict__ ws) {
    __shared__ float red[256];
    const float* zs[3]  = {z0, z1, z2};
    const float* igs[3] = {ig0, ig1, ig2};
    const int res[3] = {32, 64, 128};
    int t = blockIdx.x;
    const float* src; int n; float* dst;
    if (t < 3072) {                       // z pools: 3 levels * 16 b * 64 c
        int l = t >> 10;
        int idx = t & 1023;
        int b = idx >> 6, c = idx & 63;
        int r = res[l]; n = r * r;
        src = zs[l] + (size_t)(b * N_CH + c) * n;
        dst = ws + b * 192 + l * 64 + c;  // pooled part of ctx
    } else {                              // ig pools: 3 levels * 16 b
        int t2 = t - 3072;
        int l = t2 >> 4, b = t2 & 15;
        int r = res[l]; n = r * r;
        src = igs[l] + (size_t)b * n;
        dst = ws + 3072 + b * 3 + l;      // ig_stats
    }
    float s = 0.f;
    for (int i = threadIdx.x; i < n; i += 256) s += src[i];
    red[threadIdx.x] = s;
    __syncthreads();
    for (int off = 128; off > 0; off >>= 1) {
        if (threadIdx.x < off) red[threadIdx.x] += red[threadIdx.x + off];
        __syncthreads();
    }
    if (threadIdx.x == 0) *dst = red[0] / (float)n;
}

// ---------------- Kernel 2: MLP (WMMA) + coef/vec/avg_steps ----------------
__global__ __launch_bounds__(512)
void mlp_kernel(const float* __restrict__ ws, float* __restrict__ ws_vec,
                const float* __restrict__ current_probs, const float* __restrict__ target_vec,
                const int* __restrict__ active_attrs,
                const float* __restrict__ W1, const float* __restrict__ b1,
                const float* __restrict__ W2, const float* __restrict__ b2,
                const float* __restrict__ directions, const float* __restrict__ head_w,
                const float* __restrict__ head_b, float* __restrict__ out_avg) {
    __shared__ float ctx_s[N_B][IN_PAD];
    __shared__ float h_s[N_B][H1];
    __shared__ float c2_s[N_B][H2];
    __shared__ float bstep_s[N_B][NHEAD];
    __shared__ float coef_s[N_B][N_ATTR];
    __shared__ float hasany_s[N_ATTR];
    __shared__ int   aa_s[N_ATTR];
    __shared__ float cnt_s;

    const int tid = threadIdx.x;
    const int wave = tid >> 5, lane = tid & 31;
    const int row = lane & 15;
    const int half = lane >> 4;

    // Build ctx = [pooled0|pooled1|pooled2|current_probs|target_vec|ig_stats|0-pad]
    for (int i = tid; i < N_B * IN_PAD; i += 512) {
        int b = i / IN_PAD, j = i - b * IN_PAD;
        float v;
        if (j < 192)       v = ws[b * 192 + j];
        else if (j < 204)  v = current_probs[b * N_ATTR + (j - 192)];
        else if (j < 216)  v = target_vec[b * N_ATTR + (j - 204)];
        else if (j < 219)  v = ws[3072 + b * 3 + (j - 216)];
        else               v = 0.f;
        ctx_s[b][j] = v;
    }
    if (tid < N_ATTR) aa_s[tid] = active_attrs[tid];
    __syncthreads();

    // GEMM1: h(16x256) = relu(ctx(16x219) @ W1^T + b1); 16 waves, one N-tile each
    {
        const int nt = wave;
        v8f acc = {};
        for (int kc = 0; kc < 7; ++kc) {
            v16h af, bf;
            const int kb = kc * 32;
#pragma unroll
            for (int e = 0; e < 16; ++e) {
                int k = kb + wmma_k_idx(lane, e);
                af[e] = (_Float16)ctx_s[row][k];
                bf[e] = (k < IN_DIM) ? (_Float16)W1[(nt * 16 + row) * IN_DIM + k]
                                     : (_Float16)0.f;
            }
            acc = __builtin_amdgcn_wmma_f32_16x16x32_f16(false, af, false, bf,
                                                         (short)0, acc, false, false);
        }
#pragma unroll
        for (int v = 0; v < 8; ++v) {
            int m = v + half * 8;
            int n = nt * 16 + row;
            float val = acc[v] + b1[n];
            h_s[m][n] = val > 0.f ? val : 0.f;
        }
    }
    __syncthreads();

    // GEMM2: ctx2(16x128) = relu(h @ W2^T + b2); 8 waves
    if (wave < 8) {
        const int nt = wave;
        v8f acc = {};
        for (int kc = 0; kc < 8; ++kc) {
            v16h af, bf;
            const int kb = kc * 32;
#pragma unroll
            for (int e = 0; e < 16; ++e) {
                int k = kb + wmma_k_idx(lane, e);
                af[e] = (_Float16)h_s[row][k];
                bf[e] = (_Float16)W2[(nt * 16 + row) * H1 + k];
            }
            acc = __builtin_amdgcn_wmma_f32_16x16x32_f16(false, af, false, bf,
                                                         (short)0, acc, false, false);
        }
#pragma unroll
        for (int v = 0; v < 8; ++v) {
            int m = v + half * 8;
            int n = nt * 16 + row;
            float val = acc[v] + b2[n];
            c2_s[m][n] = val > 0.f ? val : 0.f;
        }
    }
    __syncthreads();

    // GEMM3: logits(16x36) = ctx2 @ head_w^T + head_b ; base_step = sigmoid*15+5
    if (wave < 3) {
        const int nt = wave;
        v8f acc = {};
        for (int kc = 0; kc < 4; ++kc) {
            v16h af, bf;
            const int kb = kc * 32;
#pragma unroll
            for (int e = 0; e < 16; ++e) {
                int k = kb + wmma_k_idx(lane, e);
                af[e] = (_Float16)c2_s[row][k];
                int n = nt * 16 + row;
                bf[e] = (n < NHEAD) ? (_Float16)head_w[n * H2 + k] : (_Float16)0.f;
            }
            acc = __builtin_amdgcn_wmma_f32_16x16x32_f16(false, af, false, bf,
                                                         (short)0, acc, false, false);
        }
        int n = nt * 16 + row;
        if (n < NHEAD) {
#pragma unroll
            for (int v = 0; v < 8; ++v) {
                int m = v + half * 8;
                float logit = acc[v] + head_b[n];
                float sg = 1.f / (1.f + __expf(-logit));
                bstep_s[m][n] = sg * 15.f + 5.f;
            }
        }
    }
    __syncthreads();

    // coef / has_any / count
    if (tid < N_ATTR) {
        int aa = aa_s[tid];
        float any = 0.f;
        for (int b = 0; b < N_B; ++b) {
            float tv = target_vec[b * N_ATTR + aa];
            float cp = current_probs[b * N_ATTR + aa];
            float mask = (tv != rintf(cp)) ? 1.f : 0.f;
            float d = tv - cp;
            float sg = (d > 0.f) ? 1.f : ((d < 0.f) ? -1.f : 0.f);
            coef_s[b][tid] = sg * mask;
            any = fmaxf(any, mask);
        }
        hasany_s[tid] = any;
    }
    __syncthreads();
    if (tid == 0) {
        float c = 0.f;
        for (int j = 0; j < N_ATTR; ++j) c += hasany_s[j];
        cnt_s = c;
    }
    __syncthreads();

    // vec[l][b][c] = sum_j coef[b][j] * bstep[b][aa_j][l] * directions[aa_j][c]
    for (int idx = tid; idx < N_LEV * N_B * N_CH; idx += 512) {
        int l = idx / (N_B * N_CH);
        int rem = idx - l * (N_B * N_CH);
        int b = rem >> 6, c = rem & 63;
        float s = 0.f;
        for (int j = 0; j < N_ATTR; ++j) {
            int aa = aa_s[j];
            s += coef_s[b][j] * bstep_s[b][aa * 3 + l] * directions[aa * N_CH + c];
        }
        ws_vec[idx] = s;
    }

    // avg_steps[l]
    if (tid < N_LEV) {
        int l = tid;
        float total = 0.f;
        for (int b = 0; b < N_B; ++b)
            for (int j = 0; j < N_ATTR; ++j)
                total += bstep_s[b][aa_s[j] * 3 + l] * hasany_s[j];
        float cnt = cnt_s;
        out_avg[l] = (cnt > 0.f) ? (total / fmaxf(cnt, 1.f)) / (float)N_B : 0.f;
    }
}

// ---------------- Kernel 3: mutated = z + vec * (1+4*ig) * cam ----------------
__global__ void mutate_kernel(const float* __restrict__ z, const float* __restrict__ cam,
                              const float* __restrict__ ig, const float* __restrict__ vec,
                              float* __restrict__ out, int rr) {
    int i = blockIdx.x * blockDim.x + threadIdx.x;   // one float4 per thread
    int pix4 = rr >> 2;
    int total = N_B * N_CH * pix4;
    if (i >= total) return;
    int b = i / (N_CH * pix4);
    int rem = i - b * (N_CH * pix4);
    int c = rem / pix4;
    int p = rem - c * pix4;
    float4 zz = ((const float4*)z)[i];
    float4 cm = ((const float4*)cam)[(size_t)b * pix4 + p];
    float4 gg = ((const float4*)ig)[(size_t)b * pix4 + p];
    float v = vec[b * N_CH + c];
    float4 o;
    o.x = zz.x + v * (1.f + 4.f * gg.x) * cm.x;
    o.y = zz.y + v * (1.f + 4.f * gg.y) * cm.y;
    o.z = zz.z + v * (1.f + 4.f * gg.z) * cm.z;
    o.w = zz.w + v * (1.f + 4.f * gg.w) * cm.w;
    ((float4*)out)[i] = o;
}

extern "C" void kernel_launch(void* const* d_in, const int* in_sizes, int n_in,
                              void* d_out, int out_size, void* d_ws, size_t ws_size,
                              hipStream_t stream) {
    const float* z0   = (const float*)d_in[0];
    const float* cam0 = (const float*)d_in[1];
    const float* ig0  = (const float*)d_in[2];
    const float* z1   = (const float*)d_in[3];
    const float* cam1 = (const float*)d_in[4];
    const float* ig1  = (const float*)d_in[5];
    const float* z2   = (const float*)d_in[6];
    const float* cam2 = (const float*)d_in[7];
    const float* ig2  = (const float*)d_in[8];
    const float* target_vec    = (const float*)d_in[9];
    const float* current_probs = (const float*)d_in[10];
    const int*   active_attrs  = (const int*)d_in[11];
    const float* W1 = (const float*)d_in[12];
    const float* b1 = (const float*)d_in[13];
    const float* W2 = (const float*)d_in[14];
    const float* b2 = (const float*)d_in[15];
    const float* directions = (const float*)d_in[16];
    const float* head_w = (const float*)d_in[17];
    const float* head_b = (const float*)d_in[18];

    float* out = (float*)d_out;
    float* ws  = (float*)d_ws;
    float* ws_vec = ws + 4096;                 // 3*16*64 floats

    // 1) pooling: 3072 z-pool blocks + 48 ig-pool blocks
    pool_kernel<<<3120, 256, 0, stream>>>(z0, z1, z2, ig0, ig1, ig2, ws);

    // 2) MLP (WMMA) + vec + avg_steps (single workgroup, 16 waves)
    mlp_kernel<<<1, 512, 0, stream>>>(ws, ws_vec, current_probs, target_vec, active_attrs,
                                      W1, b1, W2, b2, directions, head_w, head_b,
                                      out + 22020096);

    // 3) elementwise mutation (float4 vectorized, bandwidth bound)
    {
        int t0 = N_B * N_CH * (32 * 32) / 4;
        int t1 = N_B * N_CH * (64 * 64) / 4;
        int t2 = N_B * N_CH * (128 * 128) / 4;
        mutate_kernel<<<(t0 + 255) / 256, 256, 0, stream>>>(z0, cam0, ig0, ws_vec,        out,           32 * 32);
        mutate_kernel<<<(t1 + 255) / 256, 256, 0, stream>>>(z1, cam1, ig1, ws_vec + 1024, out + 1048576, 64 * 64);
        mutate_kernel<<<(t2 + 255) / 256, 256, 0, stream>>>(z2, cam2, ig2, ws_vec + 2048, out + 5242880, 128 * 128);
    }
}
